// SequencingDecoder_24283745092070
// MI455X (gfx1250) — compile-verified, matrix operand order checked by reading, not run on hardware
//
#include <hip/hip_runtime.h>
#include <hip/hip_bf16.h>
#include <math.h>

// ---------------------------------------------------------------------------
// Problem constants (from reference): B=64, N_ROBOTS=32, M=1024, D=512, K=32
// ---------------------------------------------------------------------------
#define BX   64
#define NR   32
#define MX   1024
#define DX   512
#define KX   32
#define RX   (BX * NR)          // 2048 rows
#define TD3  (3 * DX)           // 1536 gate outputs per weight matrix

typedef __attribute__((ext_vector_type(16))) _Float16 v16h;  // 8 VGPRs: WMMA A/B frag
typedef __attribute__((ext_vector_type(8)))  _Float16 v8h;   // 4 VGPRs: b128 chunk
typedef __attribute__((ext_vector_type(8)))  float    v8f;   // 8 VGPRs: WMMA C/D frag

// ---------------------------------------------------------------------------
// Workspace layout (bytes):
//   [0)        task_ids  : RX*KX int32     = 262144
//   [OFF_WIH)  W_ih fp16 : TD3*DX halves   = 1572864
//   [OFF_WHH)  W_hh fp16 : TD3*DX halves   = 1572864
//   [OFF_CAND) cand fp16 : RX*KX*DX halves = 67108864  (L2-resident)
// ---------------------------------------------------------------------------
#define OFF_TID  0
#define OFF_WIH  (262144)
#define OFF_WHH  (OFF_WIH + TD3 * DX * 2)
#define OFF_CAND (OFF_WHH + TD3 * DX * 2)

// ---------------- Kernel 1: fp32 -> fp16 weight downconvert ----------------
__global__ void cvt_weights_kernel(const float* __restrict__ Wih,
                                   const float* __restrict__ Whh,
                                   _Float16* __restrict__ WihH,
                                   _Float16* __restrict__ WhhH) {
    int i = blockIdx.x * blockDim.x + threadIdx.x;
    if (i < TD3 * DX) {
        WihH[i] = (_Float16)Wih[i];
        WhhH[i] = (_Float16)Whh[i];
    }
}

// ---------------- Kernel 2: stable argsort of assignment -> task_ids -------
__global__ void build_task_ids_kernel(const int* __restrict__ assignment,
                                      int* __restrict__ task_ids) {
    int b = blockIdx.x;
    int n = threadIdx.x;                    // 0..NR-1
    const int* a = assignment + (size_t)b * MX;
    int cnt = 0;
    for (int j = 0; j < MX; ++j) {
        if (a[j] == n) {
            task_ids[((size_t)b * NR + n) * KX + cnt] = j;
            ++cnt;
        }
    }
}

// ---------------- Kernel 3: gather candidates, downconvert to fp16 ---------
__global__ void gather_cands_kernel(const float* __restrict__ h_all,
                                    const int* __restrict__ task_ids,
                                    _Float16* __restrict__ cand) {
    int rk = blockIdx.x;                    // r*KX + k
    int r  = rk / KX;
    int b  = r / NR;
    int t  = task_ids[rk];
    const float* src = h_all + ((size_t)b * (NR + MX) + NR + t) * DX;
    _Float16*    dst = cand + (size_t)rk * DX;
    for (int d = threadIdx.x; d < DX; d += blockDim.x)
        dst[d] = (_Float16)src[d];
}

// ---------------- Kernel 4: persistent per-row-tile decoder -----------------
// 128 workgroups x 512 threads (16 waves). Each WG owns 16 rows for all 32
// steps. Each wave owns 2 column-tiles (16 cols each) of the hidden dim and
// accumulates 6 gate blocks (ir,iz,in,hr,hz,hn) -> gate math is wave-local.
__global__ __launch_bounds__(512, 1)
void decoder_kernel(const float* __restrict__ h_all,
                    const float* __restrict__ start_token,
                    const float* __restrict__ b_ih,
                    const float* __restrict__ b_hh,
                    const _Float16* __restrict__ WihH,
                    const _Float16* __restrict__ WhhH,
                    const _Float16* __restrict__ cand,
                    const int* __restrict__ task_ids,
                    int*   __restrict__ seq_out,
                    float* __restrict__ lp_out) {
    __shared__ __align__(32) _Float16 inp_h[16 * DX];   // GRU input x  (fp16 A operand)
    __shared__ __align__(32) _Float16 hid_h[16 * DX];   // hidden state (fp16 A operand)
    __shared__ float    scores_s[16 * KX];
    __shared__ unsigned rem_s[16];
    __shared__ int      chosen_s[16];
    __shared__ int      zoff_s;             // always 0; defeats LICM of W loads

    const int tid  = threadIdx.x;
    const int wave = tid >> 5;              // 0..15
    const int lane = tid & 31;
    const int r0   = blockIdx.x * 16;       // first global row of this tile

    // ---- init: hidden = h_robots, inp = start_token, rem = all ones -------
    {
        int row = tid >> 5;                 // 16 rows
        int d0  = (tid & 31) * 16;          // 16 contiguous dims per thread
        int r   = r0 + row;
        int b   = r / NR;
        int n   = r % NR;
        const float* hsrc = h_all + ((size_t)b * (NR + MX) + n) * DX;
        #pragma unroll
        for (int i = 0; i < 16; ++i) {
            hid_h[row * DX + d0 + i] = (_Float16)hsrc[d0 + i];
            inp_h[row * DX + d0 + i] = (_Float16)start_token[d0 + i];
        }
        if (tid < 16) rem_s[tid] = 0xFFFFFFFFu;
        if (tid == 0) zoff_s = 0;
    }

    const int   aRow = lane & 15;           // A-frag: row within 16-row tile
    const int   aKo  = (lane >> 4) << 4;    // A-frag: K offset 0 or 16
    const float inv_sqrt_d = 0.04419417382415922f;  // 1/sqrt(512)

    // Bias values per accumulator (depend only on the owned column): preload.
    float bias[2][6];
    #pragma unroll
    for (int t = 0; t < 2; ++t) {
        int col = (wave * 2 + t) * 16 + (lane & 15);
        bias[t][0] = b_ih[col];
        bias[t][1] = b_ih[DX + col];
        bias[t][2] = b_ih[2 * DX + col];
        bias[t][3] = b_hh[col];
        bias[t][4] = b_hh[DX + col];
        bias[t][5] = b_hh[2 * DX + col];
    }

    for (int step = 0; step < KX; ++step) {
        __syncthreads();

        // Opaque zero: W addresses become step-variant, so the compiler can't
        // hoist 6KB/lane of weight fragments out of the step loop (which
        // previously spilled to scratch). Weights stream from L2 instead.
        const int zoff = *(volatile int*)&zoff_s;

        // ================= GRU GEMMs on WMMA =================
        // gx = x @ W_ih^T + b_ih ; gh = h @ W_hh^T + b_hh, tiled 16x16.
        v8f acc[2][6];
        #pragma unroll
        for (int t = 0; t < 2; ++t)
            #pragma unroll
            for (int g = 0; g < 6; ++g)
                #pragma unroll
                for (int j = 0; j < 8; ++j)
                    acc[t][g][j] = bias[t][g];

        for (int kk = 0; kk < DX; kk += 32) {
            // A fragments (16x32 f16): 16 contiguous halves per lane from LDS
            const v16h a_x = *(const v16h*)(&inp_h[aRow * DX + kk + aKo]);
            const v16h a_h = *(const v16h*)(&hid_h[aRow * DX + kk + aKo]);
            #pragma unroll
            for (int t = 0; t < 2; ++t) {
                int col = (wave * 2 + t) * 16 + (lane & 15);
                // B = W^T: lane holds 16 contiguous K for column `col`,
                // K-offset (lane>>4)*16 -> contiguous 32B from row-major W.
                size_t wo = (size_t)col * DX + kk + aKo + zoff;
                const v16h b_ir = *(const v16h*)(WihH + wo);
                const v16h b_iz = *(const v16h*)(WihH + (size_t)DX * DX + wo);
                const v16h b_in = *(const v16h*)(WihH + (size_t)2 * DX * DX + wo);
                const v16h b_hr = *(const v16h*)(WhhH + wo);
                const v16h b_hz = *(const v16h*)(WhhH + (size_t)DX * DX + wo);
                const v16h b_hn = *(const v16h*)(WhhH + (size_t)2 * DX * DX + wo);
                acc[t][0] = __builtin_amdgcn_wmma_f32_16x16x32_f16(false, a_x, false, b_ir, (short)0, acc[t][0], false, false);
                acc[t][1] = __builtin_amdgcn_wmma_f32_16x16x32_f16(false, a_x, false, b_iz, (short)0, acc[t][1], false, false);
                acc[t][2] = __builtin_amdgcn_wmma_f32_16x16x32_f16(false, a_x, false, b_in, (short)0, acc[t][2], false, false);
                acc[t][3] = __builtin_amdgcn_wmma_f32_16x16x32_f16(false, a_h, false, b_hr, (short)0, acc[t][3], false, false);
                acc[t][4] = __builtin_amdgcn_wmma_f32_16x16x32_f16(false, a_h, false, b_hz, (short)0, acc[t][4], false, false);
                acc[t][5] = __builtin_amdgcn_wmma_f32_16x16x32_f16(false, a_h, false, b_hn, (short)0, acc[t][5], false, false);
            }
        }

        __syncthreads();   // all waves done reading inp_h / hid_h

        // ================= gate math (wave-local) + hidden update ==========
        #pragma unroll
        for (int t = 0; t < 2; ++t) {
            int col = (wave * 2 + t) * 16 + (lane & 15);
            #pragma unroll
            for (int j = 0; j < 8; ++j) {
                int mrow = j + ((lane >> 4) << 3);     // C/D layout row
                float r = 1.0f / (1.0f + __expf(-(acc[t][0][j] + acc[t][3][j])));
                float z = 1.0f / (1.0f + __expf(-(acc[t][1][j] + acc[t][4][j])));
                float n = tanhf(acc[t][2][j] + r * acc[t][5][j]);
                float hold = (float)hid_h[mrow * DX + col];
                float hnew = (1.0f - z) * n + z * hold;
                hid_h[mrow * DX + col] = (_Float16)hnew;
            }
        }

        __syncthreads();   // new hidden visible

        // ================= attention scores ================================
        // 512 threads == 16 rows x 32 candidates: one dot(512) per thread,
        // vectorized as 8-half chunks (b128 global + ds_load_b128).
        {
            int row = tid >> 5;
            int c   = tid & 31;
            const _Float16* cp = cand + ((size_t)(r0 + row) * KX + c) * DX;
            float s = 0.0f;
            for (int d = 0; d < DX; d += 8) {
                const v8h cv = *(const v8h*)(cp + d);
                const v8h hv = *(const v8h*)(&hid_h[row * DX + d]);
                #pragma unroll
                for (int i = 0; i < 8; ++i)
                    s += (float)cv[i] * (float)hv[i];
            }
            scores_s[row * KX + c] = s * inv_sqrt_d;
        }

        __syncthreads();

        // ================= masked softmax / argmax: wave == row ============
        {
            unsigned rm    = rem_s[wave];
            bool     alive = (rm >> lane) & 1u;
            float s  = scores_s[wave * KX + lane];
            float sm = alive ? s : -INFINITY;

            float mv = sm;
            int   mi = lane;
            #pragma unroll
            for (int off = 16; off > 0; off >>= 1) {
                float ov = __shfl_xor(mv, off, 32);
                int   oi = __shfl_xor(mi, off, 32);
                if (ov > mv || (ov == mv && oi < mi)) { mv = ov; mi = oi; }
            }
            float e = alive ? __expf(sm - mv) : 0.0f;
            float sum = e;
            #pragma unroll
            for (int off = 16; off > 0; off >>= 1) sum += __shfl_xor(sum, off, 32);
            float p  = e / sum;
            float cl = alive ? fmaxf(p, 1e-8f) : 0.0f;
            float sum2 = cl;
            #pragma unroll
            for (int off = 16; off > 0; off >>= 1) sum2 += __shfl_xor(sum2, off, 32);
            float pc = __shfl(cl, mi, 32) / sum2;
            float lp = __logf(pc + 1e-8f);

            if (lane == 0) {
                rem_s[wave]    = rm & ~(1u << mi);
                chosen_s[wave] = mi;
                int r = r0 + wave;
                seq_out[(size_t)r * KX + step] = task_ids[(size_t)r * KX + mi];
                lp_out[(size_t)r * KX + step]  = lp;
            }
        }

        __syncthreads();

        // ================= feed chosen candidate back as next input ========
        {
            int row = tid >> 5;
            int d0  = (tid & 31) * 16;
            int ch  = chosen_s[row];
            const _Float16* src = cand + ((size_t)(r0 + row) * KX + ch) * DX + d0;
            #pragma unroll
            for (int i = 0; i < 16; ++i)
                inp_h[row * DX + d0 + i] = src[i];
        }
    }
}

// ---------------------------------------------------------------------------
extern "C" void kernel_launch(void* const* d_in, const int* in_sizes, int n_in,
                              void* d_out, int out_size, void* d_ws, size_t ws_size,
                              hipStream_t stream) {
    // setup_inputs order: h_all, assignment, N, start_token, W_ih, W_hh, b_ih, b_hh
    const float* h_all       = (const float*)d_in[0];
    const int*   assignment  = (const int*)d_in[1];
    const float* start_token = (const float*)d_in[3];
    const float* W_ih        = (const float*)d_in[4];
    const float* W_hh        = (const float*)d_in[5];
    const float* b_ih        = (const float*)d_in[6];
    const float* b_hh        = (const float*)d_in[7];

    char* ws = (char*)d_ws;
    int*      task_ids = (int*)(ws + OFF_TID);
    _Float16* WihH     = (_Float16*)(ws + OFF_WIH);
    _Float16* WhhH     = (_Float16*)(ws + OFF_WHH);
    _Float16* candH    = (_Float16*)(ws + OFF_CAND);

    // Output: tuple (seq int32 [B,N,K], log_probs float32 [B,N,K]) concatenated.
    int*   seq_out = (int*)d_out;
    float* lp_out  = (float*)d_out + (size_t)RX * KX;

    // 1) weights -> fp16 (L2-resident thereafter)
    {
        int n = TD3 * DX;
        cvt_weights_kernel<<<(n + 255) / 256, 256, 0, stream>>>(W_ih, W_hh, WihH, WhhH);
    }
    // 2) stable argsort of assignment
    build_task_ids_kernel<<<BX, NR, 0, stream>>>(assignment, task_ids);
    // 3) candidate gather + fp16 downconvert (67 MB, fits in 192 MB L2)
    gather_cands_kernel<<<RX * KX, 256, 0, stream>>>(h_all, task_ids, candH);
    // 4) persistent decoder: 128 WGs x 512 threads, 16 rows/WG, 32 steps
    decoder_kernel<<<RX / 16, 512, 0, stream>>>(h_all, start_token, b_ih, b_hh,
                                                WihH, WhhH, candH, task_ids,
                                                seq_out, lp_out);
}
